// RNNFwd_9500467659440
// MI455X (gfx1250) — matrix operand order repaired and checked by reading, not compiled, not run on hardware
//
#include <hip/hip_runtime.h>

// ---- CDNA5 (gfx1250) types for WMMA ----
typedef __attribute__((ext_vector_type(16))) __bf16 bf16x16;
typedef __attribute__((ext_vector_type(8)))  __bf16 bf16x8;
typedef __attribute__((ext_vector_type(8)))  float  f32x8;
typedef __attribute__((ext_vector_type(4)))  float  f32x4;

#define DDIM   256
#define TSTEPS 2048
#define BSZ    128
#define MROWS  16          // batch rows per workgroup (WMMA "N" dim)
#define KCH    8           // 256 / 32 K-chunks per 16x16x32 bf16 WMMA
#define RS     264         // padded LDS row stride (132 DW = 4 mod 64 banks)

__device__ __forceinline__ bf16x16 cat8(bf16x8 lo, bf16x8 hi) {
    return __builtin_shufflevector(lo, hi, 0,1,2,3,4,5,6,7,8,9,10,11,12,13,14,15);
}

__device__ __forceinline__ float fast_tanh(float v) {
#if __has_builtin(__builtin_amdgcn_tanhf)
    return __builtin_amdgcn_tanhf(v);           // v_tanh_f32 (1 TRANS op)
#elif __has_builtin(__builtin_amdgcn_tanh_f32)
    return __builtin_amdgcn_tanh_f32(v);
#else
    v = fminf(fmaxf(v, -9.0f), 9.0f);
    float e = __expf(2.0f * v);                 // v_exp_f32
    return (e - 1.0f) * __builtin_amdgcn_rcpf(e + 1.0f);
#endif
}

// B-operand tile (32x16 bf16) from padded row-major LDS (two 8-bf16 runs/lane, k=8h.. and 16+8h..).
__device__ __forceinline__ bf16x16 load_btile(const __bf16* base, int c) {
    bf16x8 lo = *(const bf16x8*)(base + 32 * c);
    bf16x8 hi = *(const bf16x8*)(base + 32 * c + 16);
    return cat8(lo, hi);
}

__device__ __forceinline__ bf16x8 cvt8(f32x4 a0, f32x4 a1) {
    bf16x8 v;
    #pragma unroll
    for (int e = 0; e < 4; ++e) { v[e] = (__bf16)a0[e]; v[4 + e] = (__bf16)a1[e]; }
    return v;
}

// One timestep. All LDS pointers are loop-invariant (compile-time buffer pick).
__device__ __forceinline__ void rnn_step(
    f32x8& accX, bf16x8& xn,
    const __bf16* __restrict__ hRd,  __bf16* __restrict__ hWr,
    const __bf16* __restrict__ xRd,  __bf16* __restrict__ xWr,
    const bf16x16* __restrict__ lamA, const bf16x16* __restrict__ bA,
    float*& po, const float* __restrict__ pxb, unsigned& xoff)
{
    // ---- critical path: acc = accX + h_{t-1} @ Lam^T (8 chained WMMAs) ----
    f32x8 acc = accX;
    #pragma unroll
    for (int c = 0; c < KCH; ++c)
        acc = __builtin_amdgcn_wmma_f32_16x16x32_bf16(
                false, lamA[c], false, load_btile(hRd, c), (short)0, acc, false, false);

    // D layout: lane holds batch row j=l%16; VGPR i -> 8 consecutive features.
    float hv[8];
    #pragma unroll
    for (int i = 0; i < 8; ++i) hv[i] = fast_tanh(acc[i]);

    {   // h_t -> LDS: one 16B store per lane
        bf16x8 vh;
        #pragma unroll
        for (int i = 0; i < 8; ++i) vh[i] = (__bf16)hv[i];
        *(bf16x8*)hWr = vh;
    }
    *(bf16x8*)xWr = xn;   // stage x_{t+1} (garbage at final step: never read)

    {   // fp32 output: two 16B non-temporal stores per lane; po += DDIM
        f32x4 o0 = { hv[0], hv[1], hv[2], hv[3] };
        f32x4 o1 = { hv[4], hv[5], hv[6], hv[7] };
        __builtin_nontemporal_store(o0, (f32x4*)po);
        __builtin_nontemporal_store(o1, (f32x4*)(po + 4));
        po += DDIM;
    }

    __syncthreads();   // single barrier: publishes h_t and x_{t+1}

    // ---- off-critical: accX = x_{t+1} @ B^T ; prefetch x_{t+2} (clamped) ----
    f32x8 ax = {};
    #pragma unroll
    for (int c = 0; c < KCH; ++c)
        ax = __builtin_amdgcn_wmma_f32_16x16x32_bf16(
                false, bA[c], false, load_btile(xRd, c), (short)0, ax, false, false);
    accX = ax;

    {
        const float* px = pxb + xoff;
        f32x4 a0 = __builtin_nontemporal_load((const f32x4*)(px + 0));
        f32x4 a1 = __builtin_nontemporal_load((const f32x4*)(px + 4));
        xn = cvt8(a0, a1);
        unsigned nx = xoff + DDIM;                       // uniform scalar, clamped
        xoff = nx < (TSTEPS - 1u) * DDIM ? nx : (TSTEPS - 1u) * DDIM;
    }
}

__global__ __launch_bounds__(512)
void rnn_fwd_wmma(const float* __restrict__ x,   // [128, 2048, 256]
                  const float* __restrict__ Bm,  // [256, 256]
                  const float* __restrict__ Lam, // [256, 256]
                  float* __restrict__ out)       // [128, 2048, 256]
{
    __shared__ __bf16 hbuf[2][MROWS * RS];
    __shared__ __bf16 xbuf[2][MROWS * RS];

    const int tid    = threadIdx.x;
    const int lane   = tid & 31;
    const int wave   = tid >> 5;          // 0..15 -> features [16*wave, 16*wave+16)
    const int b0     = blockIdx.x * MROWS;
    const int n0     = wave * 16;
    const int nn     = lane & 15;
    const int hiHalf = lane >> 4;
    const size_t TD  = (size_t)TSTEPS * DDIM;

    // Resident A-operands (weights) in VGPRs: A[i,k] = W[n0+i, k], bf16.
    bf16x16 lamA[KCH];
    bf16x16 bA[KCH];
    {
        const int koA = hiHalf * 8;
        #pragma unroll
        for (int c = 0; c < KCH; ++c) {
            const float* pl = Lam + (size_t)(n0 + nn) * DDIM + 32 * c + koA;
            const float* pb = Bm  + (size_t)(n0 + nn) * DDIM + 32 * c + koA;
            f32x4 l0 = *(const f32x4*)(pl + 0);
            f32x4 l1 = *(const f32x4*)(pl + 4);
            f32x4 l2 = *(const f32x4*)(pl + 16);
            f32x4 l3 = *(const f32x4*)(pl + 20);
            f32x4 g0 = *(const f32x4*)(pb + 0);
            f32x4 g1 = *(const f32x4*)(pb + 4);
            f32x4 g2 = *(const f32x4*)(pb + 16);
            f32x4 g3 = *(const f32x4*)(pb + 20);
            lamA[c] = cat8(cvt8(l0, l1), cvt8(l2, l3));
            bA[c]   = cat8(cvt8(g0, g1), cvt8(g2, g3));
        }
    }

    // h_{-1} = 0 (read buffer of t=0 is hbuf[0]); zero padding too.
    {
        bf16x8 z = {};
        #pragma unroll 2
        for (int i = tid; i < (MROWS * RS) / 8; i += 512)
            *(bf16x8*)(&hbuf[0][i * 8]) = z;
    }
    // Per-lane streaming bases (loop-invariant).
    const float* pxb = x + (size_t)(b0 + wave) * TD + (size_t)lane * 8;  // + t*DDIM
    float* po = out + (size_t)(b0 + nn) * TD + n0 + 8 * hiHalf;          // + t*DDIM

    // Stage x[0] into xbuf[0].
    {
        f32x4 a0 = __builtin_nontemporal_load((const f32x4*)(pxb + 0));
        f32x4 a1 = __builtin_nontemporal_load((const f32x4*)(pxb + 4));
        *(bf16x8*)(&xbuf[0][wave * RS + lane * 8]) = cvt8(a0, a1);
    }
    __syncthreads();

    const int bOff = nn * RS + 8 * hiHalf;       // B-operand per-lane base offset
    const int wOffH = nn * RS + n0 + 8 * hiHalf; // h-store per-lane offset
    const int wOffX = wave * RS + lane * 8;      // x-stage per-lane offset

    // Loop-invariant per-lane LDS pointers for both ping-pong phases.
    const __bf16* hRd0 = &hbuf[0][bOff];  __bf16* hWr1 = &hbuf[1][wOffH];
    const __bf16* hRd1 = &hbuf[1][bOff];  __bf16* hWr0 = &hbuf[0][wOffH];
    const __bf16* xRd0 = &xbuf[0][bOff];  __bf16* xWr0 = &xbuf[0][wOffX];
    const __bf16* xRd1 = &xbuf[1][bOff];  __bf16* xWr1 = &xbuf[1][wOffX];

    // accX = x_0 @ B^T ; xn = x_1 ; prefetch cursor at t=2 (clamped later).
    f32x8 accX = {};
    #pragma unroll
    for (int c = 0; c < KCH; ++c)
        accX = __builtin_amdgcn_wmma_f32_16x16x32_bf16(
                 false, bA[c], false, load_btile(xRd0, c), (short)0, accX, false, false);
    bf16x8 xn;
    {
        f32x4 a0 = __builtin_nontemporal_load((const f32x4*)(pxb + DDIM + 0));
        f32x4 a1 = __builtin_nontemporal_load((const f32x4*)(pxb + DDIM + 4));
        xn = cvt8(a0, a1);
    }
    unsigned xoff = 2 * DDIM;

    for (int t = 0; t < TSTEPS; t += 2) {
        // even step: read hbuf[0]/write hbuf[1]; stage+read xbuf[1]
        rnn_step(accX, xn, hRd0, hWr1, xRd1, xWr1, lamA, bA, po, pxb, xoff);
        // odd step:  read hbuf[1]/write hbuf[0]; stage+read xbuf[0]
        rnn_step(accX, xn, hRd1, hWr0, xRd0, xWr0, lamA, bA, po, pxb, xoff);
    }
}

extern "C" void kernel_launch(void* const* d_in, const int* in_sizes, int n_in,
                              void* d_out, int out_size, void* d_ws, size_t ws_size,
                              hipStream_t stream) {
    const float* x   = (const float*)d_in[0];   // [128, 2048, 256]
    const float* Bm  = (const float*)d_in[1];   // [256, 256]
    const float* Lam = (const float*)d_in[2];   // [256, 256]
    float* out = (float*)d_out;                 // [128, 2048, 256]
    (void)in_sizes; (void)n_in; (void)out_size; (void)d_ws; (void)ws_size;

    dim3 grid(BSZ / MROWS);   // 8 workgroups, one 16-batch-row slice each
    dim3 block(512);          // 16 waves (wave32): one per 16-feature tile
    rnn_fwd_wmma<<<grid, block, 0, stream>>>(x, Bm, Lam, out);
}
